// SARANAttentionLayer_2869038153949
// MI455X (gfx1250) — matrix operand order repaired and checked by reading, not compiled, not run on hardware
//
#include <hip/hip_runtime.h>
#include <hip/hip_bf16.h>
#include <math.h>

// ---------------------------------------------------------------------------
// Types for CDNA5 WMMA (wave32): v_wmma_f32_16x16x32_bf16
//   A: 16x32 bf16 (16 halves / lane), B: 32x16 bf16, C/D: 16x16 f32 (8 / lane)
// ---------------------------------------------------------------------------
typedef __attribute__((ext_vector_type(16))) __bf16 v16bf;
typedef __attribute__((ext_vector_type(8)))  __bf16 v8bf;
typedef __attribute__((ext_vector_type(8)))  float  v8f;

union Frag16 { v16bf v; v8bf h[2]; };

__device__ __forceinline__ v8f wmma_bf16(v16bf a, v16bf b, v8f c) {
    return __builtin_amdgcn_wmma_f32_16x16x32_bf16(
        /*neg_a=*/false, a, /*neg_b=*/false, b,
        /*c_mod=*/(short)0, c, /*reuse_a=*/false, /*reuse_b=*/false);
}

// A fragment (16x32, row-major source, leading dim = ld halves):
// lane L holds row (L&15); K = (L>>4)*8 + {0..7} and +16 (per ISA layout).
__device__ __forceinline__ v16bf load_a_frag(const __bf16* base, int ld, int lane) {
    const __bf16* p = base + (lane & 15) * ld + ((lane >> 4) << 3);
    Frag16 f;
    f.h[0] = *reinterpret_cast<const v8bf*>(p);
    f.h[1] = *reinterpret_cast<const v8bf*>(p + 16);
    return f.v;
}

// B fragment (32x16) with B[k][n] = M[n][k], M row-major with leading dim ld:
// lane L holds column n=(L&15), K = (L>>4)*8 + {0..7} and +16 (per ISA layout).
__device__ __forceinline__ v16bf load_bT_frag(const __bf16* base, int ld, int lane) {
    const __bf16* p = base + (lane & 15) * ld + ((lane >> 4) << 3);
    Frag16 f;
    f.h[0] = *reinterpret_cast<const v8bf*>(p);
    f.h[1] = *reinterpret_cast<const v8bf*>(p + 16);
    return f.v;
}

// ---------------------------------------------------------------------------
// Problem constants
// ---------------------------------------------------------------------------
#define C_EMB   768
#define SEQ_T   2048
#define BATCH_B 8
#define M_ROWS  (BATCH_B * SEQ_T)      // 16384
#define N_QKV   (3 * C_EMB)            // 2304
#define K_STEPS (C_EMB / 32)           // 24
#define SCALE_Q 0.03608439182435161f   // 768^-0.5

// ---------------------------------------------------------------------------
// fp32 -> bf16 conversion
// ---------------------------------------------------------------------------
__global__ void cvt_f32_to_bf16(const float* __restrict__ src,
                                __bf16* __restrict__ dst, int n) {
    int i = blockIdx.x * blockDim.x + threadIdx.x;
    if (i < n) dst[i] = (__bf16)src[i];
}

// ---------------------------------------------------------------------------
// Core 16x64 GEMM tile: acc[s] += A(16x768) x W[n0+s*16 .. ][768]^T
// Fully unrolled, 2-stage software pipeline with compile-time stage indices
// (no register rotation copies).
// ---------------------------------------------------------------------------
__device__ __forceinline__ void gemm_tile_16x64(
        const __bf16* __restrict__ Abase, const __bf16* __restrict__ Wbase,
        int lane, v8f acc[4]) {
    v16bf a_buf[2];
    v16bf b_buf[2][4];

    // stage 0 loads (k = 0)
    a_buf[0] = load_a_frag(Abase, C_EMB, lane);
#pragma unroll
    for (int s = 0; s < 4; ++s)
        b_buf[0][s] = load_bT_frag(Wbase + (size_t)(s * 16) * C_EMB, C_EMB, lane);

#pragma unroll
    for (int it = 0; it < K_STEPS; ++it) {
        const int cur = it & 1;
        const int nxt = cur ^ 1;
        const int kn = (it + 1) * 32;
        if (kn < C_EMB) {   // compile-time under full unroll
            a_buf[nxt] = load_a_frag(Abase + kn, C_EMB, lane);
#pragma unroll
            for (int s = 0; s < 4; ++s)
                b_buf[nxt][s] =
                    load_bT_frag(Wbase + (size_t)(s * 16) * C_EMB + kn, C_EMB, lane);
        }
#pragma unroll
        for (int s = 0; s < 4; ++s)
            acc[s] = wmma_bf16(a_buf[cur], b_buf[cur][s], acc[s]);
    }
}

// ---------------------------------------------------------------------------
// QKV projection: qkv[m, n] = sum_k xh[m,k] * w_qkv[n,k]
// One wave computes a 16x64 tile. Epilogue: q scaled by 768^-0.5 (row-major),
// k row-major, v stored TRANSPOSED as vT[b][c][t] for contiguous P@V B-frags.
// ---------------------------------------------------------------------------
__global__ __launch_bounds__(32)
void qkv_gemm_wmma(const __bf16* __restrict__ xh, const __bf16* __restrict__ wh,
                   __bf16* __restrict__ qh, __bf16* __restrict__ kh,
                   __bf16* __restrict__ vT) {
    const int lane = threadIdx.x & 31;
    const int m0 = blockIdx.x * 16;
    const int n0 = blockIdx.y * 64;

    v8f acc[4] = {v8f{0.f}, v8f{0.f}, v8f{0.f}, v8f{0.f}};
    gemm_tile_16x64(xh + (size_t)m0 * C_EMB, wh + (size_t)n0 * C_EMB, lane, acc);

#pragma unroll
    for (int s = 0; s < 4; ++s) {
#pragma unroll
        for (int j = 0; j < 8; ++j) {
            const int m = m0 + j + ((lane >> 4) << 3);
            const int n = n0 + s * 16 + (lane & 15);
            const float v = acc[s][j];
            if (n < C_EMB) {
                qh[(size_t)m * C_EMB + n] = (__bf16)(v * SCALE_Q);
            } else if (n < 2 * C_EMB) {
                kh[(size_t)m * C_EMB + (n - C_EMB)] = (__bf16)v;
            } else {
                const int b = m >> 11;          // / 2048
                const int t = m & 2047;
                const int c = n - 2 * C_EMB;
                vT[((size_t)b * C_EMB + c) * SEQ_T + t] = (__bf16)v;
            }
        }
    }
}

// ---------------------------------------------------------------------------
// Fused causal flash attention.
// Grid: (T/16, B). Block: 256 threads = 8 waves.
// Per 32-key block:
//   Phase A: each wave computes S over its 96-channel K slice (6 WMMA,
//            all 6 K fragments preloaded), partials reduced in LDS.
//   Softmax: 128 threads (8 lanes/row, 4 keys/lane) with __shfl_xor row
//            reductions run the online softmax (running m, l, alpha).
//   Phase B: each wave rescales its 16x96 fp32 O accumulators and does
//            O += P(16x32, from LDS) @ V(32x96) (6 WMMA, V preloaded).
// ---------------------------------------------------------------------------
__global__ __launch_bounds__(256)
void attn_flash_wmma(const __bf16* __restrict__ qh, const __bf16* __restrict__ kh,
                     const __bf16* __restrict__ vT, __bf16* __restrict__ oh) {
    __shared__ float  Spart[8][16][32];   // per-wave score partials
    __shared__ __bf16 Pbuf[16][32];       // probabilities for P@V
    __shared__ float  mrow[16], lrow[16], arow[16];

    const int tid  = threadIdx.x;
    const int lane = tid & 31;
    const int w    = tid >> 5;            // wave id 0..7
    const int b    = blockIdx.y;
    const int t0   = blockIdx.x * 16;
    const int c0   = w * 96;              // this wave's channel slice

    if (tid < 16) { mrow[tid] = -__builtin_inff(); lrow[tid] = 0.f; }

    // Q fragments for this wave's 96 channels (kept in registers).
    v16bf qf[3];
    const __bf16* Qbase = qh + ((size_t)(b * SEQ_T + t0)) * C_EMB + c0;
#pragma unroll
    for (int kk = 0; kk < 3; ++kk)
        qf[kk] = load_a_frag(Qbase + kk * 32, C_EMB, lane);

    v8f o[6] = {v8f{0.f}, v8f{0.f}, v8f{0.f}, v8f{0.f}, v8f{0.f}, v8f{0.f}};

    __syncthreads();

    const int nkb = ((t0 + 15) >> 5) + 1;  // causal: key blocks 0..nkb-1
    for (int kb = 0; kb < nkb; ++kb) {
        const int s0 = kb * 32;

        // ---- Phase A: preload all 6 K fragments, then 6 WMMAs ----
        v16bf kf[6];
#pragma unroll
        for (int kk = 0; kk < 3; ++kk)
#pragma unroll
            for (int ns = 0; ns < 2; ++ns) {
                const __bf16* Kb = kh + ((size_t)(b * SEQ_T + s0 + ns * 16)) * C_EMB
                                      + c0 + kk * 32;
                kf[kk * 2 + ns] = load_bT_frag(Kb, C_EMB, lane);
            }
        v8f sc[2] = {v8f{0.f}, v8f{0.f}};
#pragma unroll
        for (int kk = 0; kk < 3; ++kk)
#pragma unroll
            for (int ns = 0; ns < 2; ++ns)
                sc[ns] = wmma_bf16(qf[kk], kf[kk * 2 + ns], sc[ns]);

#pragma unroll
        for (int ns = 0; ns < 2; ++ns)
#pragma unroll
            for (int j = 0; j < 8; ++j) {
                const int m = j + ((lane >> 4) << 3);
                const int n = ns * 16 + (lane & 15);
                Spart[w][m][n] = sc[ns][j];
            }
        __syncthreads();

        // ---- Online softmax: 8 lanes per row, 4 keys per lane ----
        if (tid < 128) {
            const int m   = tid >> 3;        // row 0..15
            const int sub = tid & 7;         // lane group within row
            const int tq  = t0 + m;
            float sv[4];
            float bm = -__builtin_inff();
#pragma unroll
            for (int i = 0; i < 4; ++i) {
                const int n = sub * 4 + i;
                float s = 0.f;
#pragma unroll
                for (int ww = 0; ww < 8; ++ww) s += Spart[ww][m][n];
                if (s0 + n > tq) s = -__builtin_inff();
                sv[i] = s;
                bm = fmaxf(bm, s);
            }
            // row-max across the 8 lanes of this row (stays in 8-lane group)
            bm = fmaxf(bm, __shfl_xor(bm, 1));
            bm = fmaxf(bm, __shfl_xor(bm, 2));
            bm = fmaxf(bm, __shfl_xor(bm, 4));

            const float mold = mrow[m];
            const float mnew = fmaxf(mold, bm);
            float lsum = 0.f;
#pragma unroll
            for (int i = 0; i < 4; ++i) {
                const float p = (sv[i] == -__builtin_inff())
                                    ? 0.f : __expf(sv[i] - mnew);
                lsum += p;
                Pbuf[m][sub * 4 + i] = (__bf16)p;
            }
            lsum += __shfl_xor(lsum, 1);
            lsum += __shfl_xor(lsum, 2);
            lsum += __shfl_xor(lsum, 4);
            if (sub == 0) {
                const float alpha =
                    (mold == -__builtin_inff()) ? 0.f : __expf(mold - mnew);
                mrow[m] = mnew;
                lrow[m] = lrow[m] * alpha + lsum;
                arow[m] = alpha;
            }
        }
        __syncthreads();

        // ---- Phase B: preload 6 V fragments + P, rescale O, 6 WMMAs ----
        v16bf vf[6];
#pragma unroll
        for (int ns = 0; ns < 6; ++ns) {
            const __bf16* Vb = vT + ((size_t)(b * C_EMB + c0 + ns * 16)) * SEQ_T + s0;
            vf[ns] = load_bT_frag(Vb, SEQ_T, lane);
        }
        const v16bf pf = load_a_frag(&Pbuf[0][0], 32, lane);

        // prefetch next key block's K/V lines while WMMAs run
        if (kb + 1 < nkb) {
            __builtin_prefetch(kh + ((size_t)(b * SEQ_T + s0 + 32)) * C_EMB + c0, 0, 1);
            __builtin_prefetch(vT + ((size_t)(b * C_EMB + c0)) * SEQ_T + s0 + 32, 0, 1);
        }

        const int mlo = ((lane >> 4) << 3);
#pragma unroll
        for (int ns = 0; ns < 6; ++ns) {
#pragma unroll
            for (int j = 0; j < 8; ++j) o[ns][j] *= arow[j + mlo];
            o[ns] = wmma_bf16(pf, vf[ns], o[ns]);
        }
        // next iteration's post-Phase-A barrier orders Pbuf reuse
    }

    // ---- Finalize: O /= l, store bf16 ----
#pragma unroll
    for (int ns = 0; ns < 6; ++ns)
#pragma unroll
        for (int j = 0; j < 8; ++j) {
            const int m = j + ((lane >> 4) << 3);
            const int n = c0 + ns * 16 + (lane & 15);
            const float v = o[ns][j] / lrow[m];
            oh[((size_t)(b * SEQ_T + t0 + m)) * C_EMB + n] = (__bf16)v;
        }
}

// ---------------------------------------------------------------------------
// Output projection: y[m,n] = sum_k oh[m,k] * w_out[n,k]  (fp32 output)
// ---------------------------------------------------------------------------
__global__ __launch_bounds__(32)
void out_gemm_wmma(const __bf16* __restrict__ ohh, const __bf16* __restrict__ wh,
                   float* __restrict__ y) {
    const int lane = threadIdx.x & 31;
    const int m0 = blockIdx.x * 16;
    const int n0 = blockIdx.y * 64;

    v8f acc[4] = {v8f{0.f}, v8f{0.f}, v8f{0.f}, v8f{0.f}};
    gemm_tile_16x64(ohh + (size_t)m0 * C_EMB, wh + (size_t)n0 * C_EMB, lane, acc);

#pragma unroll
    for (int s = 0; s < 4; ++s)
#pragma unroll
        for (int j = 0; j < 8; ++j) {
            const int m = m0 + j + ((lane >> 4) << 3);
            const int n = n0 + s * 16 + (lane & 15);
            y[(size_t)m * C_EMB + n] = acc[s][j];
        }
}

// ---------------------------------------------------------------------------
// Host launcher
// ---------------------------------------------------------------------------
extern "C" void kernel_launch(void* const* d_in, const int* in_sizes, int n_in,
                              void* d_out, int out_size, void* d_ws, size_t ws_size,
                              hipStream_t stream) {
    (void)n_in; (void)out_size; (void)ws_size;
    const float* x     = (const float*)d_in[0];   // [8,2048,768]
    const float* w_qkv = (const float*)d_in[1];   // [2304,768]
    const float* w_out = (const float*)d_in[2];   // [768,768]
    float* y = (float*)d_out;

    // Workspace layout (bytes)
    char* ws = (char*)d_ws;
    const size_t SZ_MC = (size_t)M_ROWS * C_EMB * sizeof(__bf16);   // 25,165,824
    __bf16* xh    = (__bf16*)(ws);
    __bf16* wqkvh = (__bf16*)(ws + SZ_MC);
    __bf16* wouth = (__bf16*)(ws + SZ_MC + 3538944);
    __bf16* qh    = (__bf16*)(ws + SZ_MC + 3538944 + 1179648);
    __bf16* kh    = (__bf16*)((char*)qh + SZ_MC);
    __bf16* vT    = (__bf16*)((char*)kh + SZ_MC);
    __bf16* oh    = (__bf16*)((char*)vT + SZ_MC);

    // 1) fp32 -> bf16 conversions
    {
        const int nx = in_sizes[0];
        cvt_f32_to_bf16<<<(nx + 255) / 256, 256, 0, stream>>>(x, xh, nx);
        const int nq = in_sizes[1];
        cvt_f32_to_bf16<<<(nq + 255) / 256, 256, 0, stream>>>(w_qkv, wqkvh, nq);
        const int no = in_sizes[2];
        cvt_f32_to_bf16<<<(no + 255) / 256, 256, 0, stream>>>(w_out, wouth, no);
    }

    // 2) QKV projection (16384 x 2304 x 768)
    {
        dim3 grid(M_ROWS / 16, N_QKV / 64);
        qkv_gemm_wmma<<<grid, 32, 0, stream>>>(xh, wqkvh, qh, kh, vT);
    }

    // 3) Fused causal flash attention
    {
        dim3 grid(SEQ_T / 16, BATCH_B);
        attn_flash_wmma<<<grid, 256, 0, stream>>>(qh, kh, vT, oh);
    }

    // 4) Output projection (16384 x 768 x 768) -> fp32
    {
        dim3 grid(M_ROWS / 16, C_EMB / 64);
        out_gemm_wmma<<<grid, 32, 0, stream>>>(oh, wouth, y);
    }
}
